// PersistenceImageTorch_48189533061627
// MI455X (gfx1250) — matrix operand order repaired and checked by reading, not compiled, not run on hardware
//
#include <hip/hip_runtime.h>

typedef __attribute__((ext_vector_type(16))) _Float16 v16h;
typedef __attribute__((ext_vector_type(8)))  _Float16 v8h;
typedef __attribute__((ext_vector_type(2)))  __fp16   fp16x2;   // matches cvt_pkrtz return type
typedef __attribute__((ext_vector_type(8)))  float    v8f;

// Problem constants (from reference): bw = 0.1, 64x64 grid on [0,1]^2, N = 2048.
#define NPTS        2048
#define NCHUNK      (NPTS / 32)            // 64 K-chunks of 32 points
#define LDS_STRIDE  40                     // halves per row (80 B): 16B-aligned, bank-spread pad
#define COEF        72.13475204444817f     // (1/(2*bw^2)) * log2(e) = 50 * 1.442695...
#define NORM        15.915494309189535f    // 1/(2*pi*bw^2)
#define GSTEP       (1.0f / 63.0f)         // linspace(0,1,64) step

__global__ __launch_bounds__(128)
void pimg_wmma_kernel(const float* __restrict__ X, float* __restrict__ out) {
    const int t  = threadIdx.x;      // 0..127
    const int b  = blockIdx.x;       // batch
    const int my = blockIdx.y;       // output row-strip 0..3 (rows y0..y0+15)
    const int wv = t >> 5;           // wave id 0..3 == output tile column mx
    const int ln = t & 31;           // lane in wave32

    // Double-buffered separated Gaussian factors, f16 in LDS (one barrier per chunk)
    __shared__ _Float16 gxs[2][64 * LDS_STRIDE];   // Gx[x][k]  x=0..63, k=0..31
    __shared__ _Float16 gys[2][16 * LDS_STRIDE];   // Gy[y-y0][k]

    const float4* Xp = (const float4*)(X + (size_t)b * (NPTS * 2)); // 2 points per float4
    const int kp = t & 15;           // point-pair index within chunk (points 2kp, 2kp+1)
    const int rb = t >> 4;           // 0..7 row base for producer phase

    const int y0 = my * 16;
    const int m    = ln & 15;
    const int half = ln >> 4;

    v8f acc = {};                    // f32 16x16 accumulator fragment

    for (int c = 0; c < NCHUNK; ++c) {
        const int pb = c & 1;

        // ----- produce: 32 points -> Gx (64x32), Gy (16x32) in f16 LDS -----
        // Safe without a back-guard barrier: by barrier(c-1), all threads had
        // finished consume(c-2), so buffer pb == (c-2)&1 is free to overwrite.
        float4 pt = Xp[c * 16 + kp];             // 16B load: points 2kp, 2kp+1
        float px0 = pt.x, py0 = pt.y - pt.x;     // (birth, persistence)
        float px1 = pt.z, py1 = pt.w - pt.z;

        #pragma unroll
        for (int p = 0; p < 8; ++p) {            // Gx: all 64 grid columns
            int r = rb + p * 8;
            float g  = (float)r * GSTEP;
            float d0 = px0 - g, d1 = px1 - g;
            float e0 = __builtin_amdgcn_exp2f(d0 * d0 * -COEF);  // v_exp_f32
            float e1 = __builtin_amdgcn_exp2f(d1 * d1 * -COEF);
            fp16x2 pk = __builtin_amdgcn_cvt_pkrtz(e0, e1);      // v_cvt_pk_rtz_f16_f32
            *(fp16x2*)(&gxs[pb][r * LDS_STRIDE + 2 * kp]) = pk;
        }
        #pragma unroll
        for (int p = 0; p < 2; ++p) {            // Gy: this block's 16 grid rows
            int r = rb + p * 8;                  // local 0..15
            float g  = (float)(y0 + r) * GSTEP;
            float d0 = py0 - g, d1 = py1 - g;
            float e0 = __builtin_amdgcn_exp2f(d0 * d0 * -COEF);
            float e1 = __builtin_amdgcn_exp2f(d1 * d1 * -COEF);
            fp16x2 pk = __builtin_amdgcn_cvt_pkrtz(e0, e1);
            *(fp16x2*)(&gys[pb][r * LDS_STRIDE + 2 * kp]) = pk;
        }
        __syncthreads();                         // single barrier per chunk

        // ----- consume: build WMMA fragments from LDS, accumulate -----
        // A = Gy^T tile (16x32 f16): lane row M=m; K pairs per ISA A-layout
        const _Float16* arow = &gys[pb][m * LDS_STRIDE];
        v8h a0 = *(const v8h*)(arow + half * 8);        // K = h*8 .. h*8+7      (VGPRs 0-3)
        v8h a1 = *(const v8h*)(arow + 16 + half * 8);   // K = 16+h*8 .. +7      (VGPRs 4-7)
        v16h A = __builtin_shufflevector(a0, a1, 0,1,2,3,4,5,6,7,8,9,10,11,12,13,14,15);

        // B = Gx tile (32x16 f16): lane col N=m; K = h*8 .. h*8+7, 16+h*8 .. +7
        const _Float16* brow = &gxs[pb][(wv * 16 + m) * LDS_STRIDE + half * 8];
        v8h b0 = *(const v8h*)(brow);
        v8h b1 = *(const v8h*)(brow + 16);
        v16h B = __builtin_shufflevector(b0, b1, 0,1,2,3,4,5,6,7,8,9,10,11,12,13,14,15);

        acc = __builtin_amdgcn_wmma_f32_16x16x32_f16(
            /*neg_a=*/false, A, /*neg_b=*/false, B,
            /*c_mod=*/(short)0, acc, /*reuse_a=*/false, /*reuse_b=*/false);
    }

    // ----- epilogue: scale by 1/(2*pi*bw^2), scatter per C-matrix VGPR layout -----
    {
        const int x = wv * 16 + m;
        float* op = out + (size_t)b * 4096;
        #pragma unroll
        for (int j = 0; j < 8; ++j) {
            int y = y0 + half * 8 + j;          // VGPR j -> M = j (+8 for lanes 16..31)
            op[y * 64 + x] = acc[j] * NORM;
        }
    }
}

extern "C" void kernel_launch(void* const* d_in, const int* in_sizes, int n_in,
                              void* d_out, int out_size, void* d_ws, size_t ws_size,
                              hipStream_t stream) {
    (void)n_in; (void)d_ws; (void)ws_size; (void)out_size;
    const float* X = (const float*)d_in[0];
    float* out = (float*)d_out;
    int B = in_sizes[0] / (NPTS * 2);           // 32 for the reference shapes
    dim3 grid(B, 4, 1);                          // (batch, 16-row output strip)
    dim3 block(128, 1, 1);                       // 4 wave32s; wave = output tile column
    pimg_wmma_kernel<<<grid, block, 0, stream>>>(X, out);
}
